// SNNPricePredictor_20993800143457
// MI455X (gfx1250) — matrix operand order repaired and checked by reading, not compiled, not run on hardware
//
#include <hip/hip_runtime.h>
#include <cmath>

typedef __attribute__((ext_vector_type(16))) _Float16 v16h;
typedef __attribute__((ext_vector_type(8)))  float    v8f;
typedef __attribute__((ext_vector_type(4)))  unsigned v4u;
typedef __attribute__((ext_vector_type(4)))  int      v4i;

#define WMMA_F16(a,b,c) __builtin_amdgcn_wmma_f32_16x16x32_f16(false,(a),false,(b),(short)0,(c),false,false)

static __device__ __forceinline__ unsigned ballot32(bool p) {
  return __builtin_amdgcn_ballot_w32(p);
}

#if __has_builtin(__builtin_amdgcn_global_load_async_to_lds_b128)
#define HAVE_ASYNC_LDS 1
typedef __attribute__((address_space(1))) v4i gv4i;   // global int4
typedef __attribute__((address_space(3))) v4i lv4i;   // LDS int4
// Async DMA 16B global -> LDS (per lane), tracked by ASYNCcnt.
static __device__ __forceinline__ void async_ld16(const void* g, void* l) {
  __builtin_amdgcn_global_load_async_to_lds_b128(
      (gv4i*)(size_t)g, (lv4i*)(unsigned)(size_t)l, 0, 0);
}
static __device__ __forceinline__ void wait_async0() {
  asm volatile("s_wait_asynccnt 0" ::: "memory");
}
#else
#define HAVE_ASYNC_LDS 0
#endif

// Problem constants (fixed by the reference)
#define NB     65536
#define NSTEP  25
#define BETA   0.95f
// spike word grid: [t][btile(4096)][i(8)][ntg(64)]  (u32 each)
#define SPK_T_STRIDE (4096u * 512u)   // words per time step

// ---------------------------------------------------------------------------
// Prep: convert W1/W2 to f16 and pack into per-lane WMMA B-tile layout.
// B-tile (32x16 f16): lane L holds col N=L%16, K_local = e<8 ? 8*(L/16)+e : 16+8*(L/16)+(e-8), e=0..15,
// stored lane-major: halfs[ (tile*32 + L)*16 + e ].
// ---------------------------------------------------------------------------
__global__ void snn_pack_weights(const float* __restrict__ W1,
                                 const float* __restrict__ W2,
                                 _Float16* __restrict__ w1p,
                                 _Float16* __restrict__ w2p) {
  int id = blockIdx.x * 256 + threadIdx.x;           // 524288 threads
  // W1: [64,1024] -> tiles (kt 0..1, nt 0..63): 65536 halfs
  if (id < 65536) {
    int e  = id & 15;
    int L  = (id >> 4) & 31;
    int nt = (id >> 9) & 63;
    int kt = id >> 15;
    int K  = 32 * kt + ((e < 8) ? (8 * (L >> 4) + e) : (16 + 8 * (L >> 4) + (e - 8)));
    int N  = 16 * nt + (L & 15);
    w1p[id] = (_Float16)W1[K * 1024 + N];
  }
  // W2: [1024,512] -> tiles (kt 0..31, nt 0..31): 524288 halfs
  if (id < 524288) {
    int e  = id & 15;
    int L  = (id >> 4) & 31;
    int nt = (id >> 9) & 31;
    int kt = id >> 14;
    int K  = 32 * kt + ((e < 8) ? (8 * (L >> 4) + e) : (16 + 8 * (L >> 4) + (e - 8)));
    int N  = 16 * nt + (L & 15);
    w2p[id] = (_Float16)W2[K * 512 + N];
  }
}

// ---------------------------------------------------------------------------
// Pass 1: cur1 = x@W1+b1 via WMMA, then 25-step leaky recurrence in registers,
// spikes bit-packed with wave ballots. One wave per 16-row batch tile.
// Chunk = 4 n-tiles (64 cols) -> acc/cm1/mem are 32 VGPRs each (no spills),
// and 32 ballots/step fill exactly one 32-lane word buffer.
// Ballot word (t, tile, i, ntg): bit (c + 16*hh) = spike(row = i + 8*hh,
// col = 16*ntg + c) of that tile.
// ---------------------------------------------------------------------------
__global__ void snn_pass1(const float* __restrict__ x,
                          const float* __restrict__ b1,
                          const _Float16* __restrict__ w1p,
                          unsigned* __restrict__ spk) {
  const int lane = threadIdx.x & 31;
  const int wid  = threadIdx.x >> 5;
  const int tile = blockIdx.x * 8 + wid;             // 0..4095
  const int b0   = tile * 16;
  const int r    = lane & 15;
  const int h    = lane >> 4;

  // Load A-tiles of x (16x32 f16 layout): element e<8 -> K=8h+e, e>=8 -> K=16+8h+(e-8)
  v16h a0, a1;
  const float* xrow = x + (size_t)(b0 + r) * 64;
  #pragma unroll
  for (int e = 0; e < 16; e++) {
    int k = (e < 8) ? (8 * h + e) : (16 + 8 * h + (e - 8));
    a0[e] = (_Float16)xrow[k];
    a1[e] = (_Float16)xrow[32 + k];
  }

  // 16 chunks of 4 n-tiles (64 cols) each; full recurrence per chunk.
  #pragma unroll 1
  for (int c = 0; c < 16; c++) {
    v8f acc[4];
    #pragma unroll
    for (int j = 0; j < 4; j++) {
      float bv = b1[16 * (4 * c + j) + r];
      v8f t = {bv, bv, bv, bv, bv, bv, bv, bv};
      acc[j] = t;
    }
    #pragma unroll
    for (int j = 0; j < 4; j++) {
      v16h bt0 = *(const v16h*)(w1p + ((size_t)((0 * 64 + 4 * c + j) * 32 + lane)) * 16);
      acc[j] = WMMA_F16(a0, bt0, acc[j]);
      v16h bt1 = *(const v16h*)(w1p + ((size_t)((1 * 64 + 4 * c + j) * 32 + lane)) * 16);
      acc[j] = WMMA_F16(a1, bt1, acc[j]);
    }
    // cur1 fixed across steps; precompute cur1-1 for the reset-fused update
    v8f mem[4], cm1[4];
    #pragma unroll
    for (int j = 0; j < 4; j++) { v8f z = {}; mem[j] = z; cm1[j] = acc[j] - 1.0f; }

    #pragma unroll 1
    for (int t = 0; t < NSTEP; t++) {
      unsigned wbuf = 0;
      #pragma unroll
      for (int i = 0; i < 8; i++) {
        #pragma unroll
        for (int j = 0; j < 4; j++) {
          bool  pv = mem[j][i] > 1.0f;                 // reset from previous mem
          float s  = pv ? cm1[j][i] : acc[j][i];       // cur1 - reset*theta
          float m  = fmaf(BETA, mem[j][i], s);
          mem[j][i] = m;
          unsigned bm = ballot32(m > 1.0f);            // spike bits across wave
          const int widx = i * 4 + j;                  // 0..31
          if (lane == widx) wbuf = bm;
        }
      }
      // lane L holds word (i = L>>2, ntg = 4c + (L&3)); one coalesced NT store
      unsigned* dst = spk + (size_t)t * SPK_T_STRIDE + (size_t)tile * 512;
      __builtin_nontemporal_store(wbuf, &dst[(lane >> 2) * 64 + 4 * c + (lane & 3)]);
    }
  }
}

// ---------------------------------------------------------------------------
// Pass 2: 25 steps of  mem2 = beta*mem2 - reset + b2 + spk1@W2  with mem2 held
// in WMMA accumulators; layer-3 masked reduction + mem3 recurrence.
// WG = 256 thr (8 waves), 64 batch rows. Wave w owns n-tiles 4w..4w+3 and all
// 4 row-tiles, so each packed-W2 B-tile is read once per WG per step.
// Spike words for step t+1 are DMA'd into the other LDS buffer with
// GLOBAL_LOAD_ASYNC_TO_LDS_B128 while step t computes.
// ---------------------------------------------------------------------------
__global__ void snn_pass2(const float* __restrict__ b2,
                          const float* __restrict__ W3,
                          const float* __restrict__ b3,
                          const _Float16* __restrict__ w2p,
                          const unsigned* __restrict__ spk,
                          float* __restrict__ out) {
  __shared__ v4u  lds_spk4[2][512];                    // double-buffered 2x8KB
  __shared__ float lds_red[64 * 8];                    // per-row, per-wave partials

  const int tid  = threadIdx.x;
  const int lane = tid & 31;
  const int wid  = tid >> 5;
  const int b0   = blockIdx.x * 64;
  const int r    = lane & 15;
  const int h    = lane >> 4;
  const int byteoff = 8 * h + 16 * (r >> 3);

  float b2c[4], b2m[4], w3c[4];
  #pragma unroll
  for (int j = 0; j < 4; j++) {
    int col = 16 * (4 * wid + j) + r;
    b2c[j] = b2[col];
    b2m[j] = b2c[j] - 1.0f;
    w3c[j] = W3[col];
  }
  const float b3v = b3[0];

  v8f acc[4][4] = {};                                  // mem2: [row-tile][n-tile]
  float mem3 = 0.0f, sum3 = 0.0f;                      // used by tid<64

  const size_t blk_off = (size_t)blockIdx.x * 2048;

#if HAVE_ASYNC_LDS
  {  // prologue: DMA step-0 spikes into buffer 0
    const v4u* src = (const v4u*)(spk + blk_off);
    async_ld16(&src[tid],       &lds_spk4[0][tid]);
    async_ld16(&src[tid + 256], &lds_spk4[0][tid + 256]);
  }
#endif

  #pragma unroll 1
  for (int t = 0; t < NSTEP; t++) {
    const int buf = t & 1;
#if HAVE_ASYNC_LDS
    wait_async0();                                     // this step's DMA done
    __syncthreads();
    if (t + 1 < NSTEP) {                               // overlap next step's DMA
      const v4u* nsrc = (const v4u*)(spk + (size_t)(t + 1) * SPK_T_STRIDE + blk_off);
      async_ld16(&nsrc[tid],       &lds_spk4[buf ^ 1][tid]);
      async_ld16(&nsrc[tid + 256], &lds_spk4[buf ^ 1][tid + 256]);
    }
#else
    {
      const v4u* src = (const v4u*)(spk + (size_t)t * SPK_T_STRIDE + blk_off);
      lds_spk4[buf][tid]       = __builtin_nontemporal_load(&src[tid]);
      lds_spk4[buf][tid + 256] = __builtin_nontemporal_load(&src[tid + 256]);
      if (t + 1 < NSTEP) {
        const char* nsrc = (const char*)(spk + (size_t)(t + 1) * SPK_T_STRIDE + blk_off);
        __builtin_prefetch(nsrc + tid * 32, 0, 1);     // warm next step's 8KB
      }
    }
    __syncthreads();
#endif
    const unsigned* lds_spk = (const unsigned*)lds_spk4[buf];

    // fuse: acc = beta*mem2_old - reset(mem2_old) + b2 ; WMMA chain adds spk1@W2
    #pragma unroll
    for (int rt = 0; rt < 4; rt++)
      #pragma unroll
      for (int j = 0; j < 4; j++)
        #pragma unroll
        for (int i = 0; i < 8; i++) {
          float s = (acc[rt][j][i] > 1.0f) ? b2m[j] : b2c[j];
          acc[rt][j][i] = fmaf(BETA, acc[rt][j][i], s);
        }

    #pragma unroll 2
    for (int kt = 0; kt < 32; kt++) {
      if (kt < 31)
        __builtin_prefetch(w2p + (size_t)(((kt + 1) * 32 + 4 * wid) * 32) * 16, 0, 3);
      // unpack 4 A-tiles (16x32 binary -> f16) from ballot words
      v16h a[4];
      #pragma unroll
      for (int rt = 0; rt < 4; rt++) {
        const unsigned* wp = &lds_spk[rt * 512 + (r & 7) * 64 + 2 * kt];
        unsigned wA = wp[0], wB = wp[1];
        unsigned ba = (wA >> byteoff) & 0xFFu;
        unsigned bb = (wB >> byteoff) & 0xFFu;
        v16h av;
        #pragma unroll
        for (int e = 0; e < 8; e++) {
          av[e]     = ((ba >> e) & 1u) ? (_Float16)1.0f : (_Float16)0.0f;
          av[e + 8] = ((bb >> e) & 1u) ? (_Float16)1.0f : (_Float16)0.0f;
        }
        a[rt] = av;
      }
      #pragma unroll
      for (int j = 0; j < 4; j++) {
        int nt = 4 * wid + j;
        v16h bt = *(const v16h*)(w2p + ((size_t)((kt * 32 + nt) * 32 + lane)) * 16);
        #pragma unroll
        for (int rt = 0; rt < 4; rt++)
          acc[rt][j] = WMMA_F16(a[rt], bt, acc[rt][j]);
      }
    }

    // layer 3: masked sum of W3 over spiking elements, reduce 16 lanes/half
    #pragma unroll
    for (int rt = 0; rt < 4; rt++) {
      #pragma unroll
      for (int i = 0; i < 8; i++) {
        float p = 0.0f;
        #pragma unroll
        for (int j = 0; j < 4; j++)
          p += (acc[rt][j][i] > 1.0f) ? w3c[j] : 0.0f;
        p += __shfl_xor(p, 1, 32);
        p += __shfl_xor(p, 2, 32);
        p += __shfl_xor(p, 4, 32);
        p += __shfl_xor(p, 8, 32);
        if (r == 0) {
          int row = rt * 16 + i + 8 * h;
          lds_red[row * 8 + wid] = p;
        }
      }
    }
    __syncthreads();

    if (tid < 64) {                                    // deterministic reduction + mem3
      float cur3 = b3v;
      #pragma unroll
      for (int w = 0; w < 8; w++) cur3 += lds_red[tid * 8 + w];
      float rs = (mem3 > 1.0f) ? 1.0f : 0.0f;
      mem3 = fmaf(BETA, mem3, cur3 - rs);
      sum3 += mem3;
    }
  }

  if (tid < 64)
    out[b0 + tid] = 1.0f / (1.0f + expf(-sum3 * (1.0f / (float)NSTEP)));
}

// ---------------------------------------------------------------------------
extern "C" void kernel_launch(void* const* d_in, const int* in_sizes, int n_in,
                              void* d_out, int out_size, void* d_ws, size_t ws_size,
                              hipStream_t stream) {
  const float* x  = (const float*)d_in[0];
  const float* W1 = (const float*)d_in[1];
  const float* b1 = (const float*)d_in[2];
  const float* W2 = (const float*)d_in[3];
  const float* b2 = (const float*)d_in[4];
  const float* W3 = (const float*)d_in[5];
  const float* b3 = (const float*)d_in[6];
  float* out = (float*)d_out;

  char* ws = (char*)d_ws;
  _Float16* w1p = (_Float16*)(ws);                 // 128 KB
  _Float16* w2p = (_Float16*)(ws + (1 << 17));     // 1 MB
  unsigned* spk = (unsigned*)(ws + (1 << 21));     // 25*4096*512 u32 = 210 MB

  snn_pack_weights<<<2048, 256, 0, stream>>>(W1, W2, w1p, w2p);
  snn_pass1<<<512, 256, 0, stream>>>(x, b1, w1p, spk);
  snn_pass2<<<1024, 256, 0, stream>>>(b2, W3, b3, w2p, spk, out);
}
